// Attention_82643760710180
// MI455X (gfx1250) — matrix operand order repaired and checked by reading, not compiled, not run on hardware
//
#include <hip/hip_runtime.h>
#include <math.h>

#define B_   64
#define S_   512
#define H_   1024
#define E_   768
#define CDIM (H_ + E_)

typedef __attribute__((ext_vector_type(16))) __bf16 v16bf;
typedef __attribute__((ext_vector_type(2)))  __bf16 v2bf;
typedef __attribute__((ext_vector_type(8)))  float  v8f;

union Frag { v16bf v; uint4 q[2]; };

#ifndef __has_builtin
#define __has_builtin(x) 0
#endif

// f32 pair -> packed bf16 dword (lo in [15:0], hi in [31:16]).
// No v_cvt_pk_bf16_f32 builtin on this toolchain for gfx1250, so use
// round-half-up + one v_perm_b32 splice of the two high halves (3 VALU total,
// all 32-bit ops -- avoids the 16-bit-register shuffle swamp).
__device__ __forceinline__ unsigned int pk2bf(float lo, float hi) {
#if __has_builtin(__builtin_amdgcn_cvt_pk_bf16_f32)
  v2bf r = __builtin_amdgcn_cvt_pk_bf16_f32(lo, hi);
  return __builtin_bit_cast(unsigned int, r);
#else
  unsigned int a = __float_as_uint(lo) + 0x8000u;
  unsigned int b = __float_as_uint(hi) + 0x8000u;
  // result bytes {b3,b2,a3,a2}: sel 0-3 -> S1 (a), 4-7 -> S0 (b)
  return __builtin_amdgcn_perm(b, a, 0x07060302u);
#endif
}

// native V_TANH_F32 on gfx1250; branch-free saturating fallback otherwise
__device__ __forceinline__ float fast_tanh(float x) {
#if __has_builtin(__builtin_amdgcn_tanhf)
  return __builtin_amdgcn_tanhf(x);
#else
  float e = __builtin_amdgcn_exp2f(x * 2.8853900817779268f); // 2*log2(e)*x
  return 1.0f - 2.0f * __builtin_amdgcn_rcpf(e + 1.0f);
#endif
}

__global__ void zero_f32(float* __restrict__ p, int n) {
  int i = blockIdx.x * 256 + threadIdx.x;
  if (i < n) p[i] = 0.0f;
}

__device__ __forceinline__ void load_tile(const float* __restrict__ p, int kt, float4 r[4]) {
#pragma unroll
  for (int j = 0; j < 4; ++j) r[j] = *(const float4*)(p + kt * 32 + j * 4);
}

__device__ __forceinline__ void stage_tile(unsigned int* dst, unsigned int rowU,
                                           const float4 r[4]) {
#pragma unroll
  for (int j = 0; j < 4; ++j) {
    dst[rowU + j * 2]     = pk2bf(r[j].x, r[j].y);
    dst[rowU + j * 2 + 1] = pk2bf(r[j].z, r[j].w);
  }
}

// C = W (R x K, row-major) * X^T (X is S x K row-major), fused:
// scores[b, s] += sum_r wc[b, wcOff + r] * tanh(C[r, s])
// 128x128 tile / workgroup, BK=32, bf16 WMMA f32-accum, ping-pong LDS
// (one barrier per K-step; staging stores overlap WMMA tail).
__global__ __launch_bounds__(256) void gemm_tanh_score(
    const float* __restrict__ W, const float* __restrict__ X,
    const float* __restrict__ wc, float* __restrict__ scores,
    int K, int nRowBlk, int nColBlk, int wcOff)
{
  // bf16 tiles, row stride padded to 40 bf16 (=20 dwords, 80B) for bank spread
  constexpr unsigned BUF = 128u * 20u;
  __shared__ unsigned int ldsA[2 * BUF];
  __shared__ unsigned int ldsB[2 * BUF];
  __shared__ float lds_wc[128];
  __shared__ float lds_sc[128];

  const int t     = threadIdx.x;
  const int lane  = t & 31;
  const int wid   = t >> 5;
  const int wm    = wid >> 1;   // 0..3 -> row offset wm*32
  const int wn    = wid & 1;    // 0..1 -> col offset wn*64
  const int lrow  = lane & 15;
  const int lhalf = lane >> 4;

  const int cb  = blockIdx.x % nColBlk;
  int       tmp = blockIdx.x / nColBlk;
  const int rb  = tmp % nRowBlk;
  const int b   = tmp / nRowBlk;
  const int R   = nRowBlk * 128;

  // cooperative tile loader mapping: 256 threads, each 16 f32 of a 128x32 tile
  const int ldRow = t >> 1;
  const int ldCol = (t & 1) * 16;

  const float* Wp = W + (size_t)b * R  * K + (size_t)(rb * 128 + ldRow) * K + ldCol;
  const float* Xp = X + (size_t)b * S_ * K + (size_t)(cb * 128 + ldRow) * K + ldCol;
  const unsigned int rowU = (unsigned)ldRow * 20u + (unsigned)(ldCol >> 1);

  // frag LDS dword offsets (within one buffer)
  unsigned int aBase[2], bBase[4];
#pragma unroll
  for (int m = 0; m < 2; ++m)
    aBase[m] = (unsigned)(wm * 32 + m * 16 + lrow) * 20u + (unsigned)lhalf * 4u;
#pragma unroll
  for (int n = 0; n < 4; ++n)
    bBase[n] = (unsigned)(wn * 64 + n * 16 + lrow) * 20u + (unsigned)lhalf * 4u;

  if (t < 128) {
    lds_wc[t] = wc[(size_t)b * CDIM + wcOff + rb * 128 + t];
    lds_sc[t] = 0.0f;
  }

  v8f acc[2][4];
  const v8f vz = {0.f, 0.f, 0.f, 0.f, 0.f, 0.f, 0.f, 0.f};
#pragma unroll
  for (int m = 0; m < 2; ++m)
#pragma unroll
    for (int n = 0; n < 4; ++n) acc[m][n] = vz;

  const int nK = K >> 5;

  // prologue: stage tile 0 into buffer 0
  float4 ar[4], br[4];
  load_tile(Wp, 0, ar);
  load_tile(Xp, 0, br);
  stage_tile(ldsA, rowU, ar);
  stage_tile(ldsB, rowU, br);
  __syncthreads();

  for (int kt = 0; kt < nK; ++kt) {
    const int cur = kt & 1;
    const int nxt = cur ^ 1;
    const bool has_next = (kt + 1) < nK;
    if (has_next) {                       // issue next global loads early
      load_tile(Wp, kt + 1, ar);
      load_tile(Xp, kt + 1, br);
    }

    const unsigned int* bufA = ldsA + cur * BUF;
    const unsigned int* bufB = ldsB + cur * BUF;

    Frag a[2], bf[4];
    // A frag (16x32 bf16): VGPR0-3 = K0..7 (lanes0-15) / K8..15 (lanes16-31),
    //                      VGPR4-7 = K16..23 / K24..31
#pragma unroll
    for (int m = 0; m < 2; ++m) {
      a[m].q[0] = *(const uint4*)(bufA + aBase[m]);      // +0B
      a[m].q[1] = *(const uint4*)(bufA + aBase[m] + 8);  // +32B
    }
    // B frag (32x16 bf16): VGPR0-3 = K0..7 (lanes0-15) / K8..15 (lanes16-31), VGPR4-7 = K16..23 / K24..31
#pragma unroll
    for (int n = 0; n < 4; ++n) {
      bf[n].q[0] = *(const uint4*)(bufB + bBase[n]);      // +0B
      bf[n].q[1] = *(const uint4*)(bufB + bBase[n] + 8);  // +32B
    }
#pragma unroll
    for (int m = 0; m < 2; ++m)
#pragma unroll
      for (int n = 0; n < 4; ++n)
        acc[m][n] = __builtin_amdgcn_wmma_f32_16x16x32_bf16(
            false, a[m].v, false, bf[n].v, (short)0, acc[m][n], false, false);

    if (has_next) {                       // overlap staging with WMMA tail
      stage_tile(ldsA + nxt * BUF, rowU, ar);
      stage_tile(ldsB + nxt * BUF, rowU, br);
    }
    __syncthreads();  // next iter's reads see the stores; frag loads of this
                      // iter were dscnt-waited before the WMMAs issued
  }

  // Epilogue: tanh, dot with w_combine rows, reduce into per-column scores
#pragma unroll
  for (int n = 0; n < 4; ++n) {
    float ssum = 0.0f;
#pragma unroll
    for (int m = 0; m < 2; ++m) {
#pragma unroll
      for (int i = 0; i < 8; ++i) {
        const int r = wm * 32 + m * 16 + (lhalf << 3) + i;  // C/D layout: VGPR i -> M = i or 8+i
        ssum += lds_wc[r] * fast_tanh(acc[m][n][i]);
      }
    }
    atomicAdd(&lds_sc[wn * 64 + n * 16 + lrow], ssum);
  }
  __syncthreads();
  if (t < 128)
    unsafeAtomicAdd(&scores[(size_t)b * S_ + cb * 128 + t], lds_sc[t]);
}

// per-batch softmax over S=512 -> weight
__global__ __launch_bounds__(256) void softmax_k(const float* __restrict__ scores,
                                                 float* __restrict__ weight)
{
  const int b = blockIdx.x, t = threadIdx.x;
  __shared__ float red[256];
  float v0 = scores[b * S_ + t];
  float v1 = scores[b * S_ + 256 + t];
  red[t] = fmaxf(v0, v1);
  __syncthreads();
  for (int s = 128; s > 0; s >>= 1) {
    if (t < s) red[t] = fmaxf(red[t], red[t + s]);
    __syncthreads();
  }
  const float m = red[0];
  __syncthreads();
  float e0 = expf(v0 - m), e1 = expf(v1 - m);
  red[t] = e0 + e1;
  __syncthreads();
  for (int s = 128; s > 0; s >>= 1) {
    if (t < s) red[t] += red[t + s];
    __syncthreads();
  }
  const float inv = 1.0f / red[0];
  weight[b * S_ + t]       = e0 * inv;
  weight[b * S_ + 256 + t] = e1 * inv;
}

// out[b, h] = sum_s text[b, s, h] * weight[b, s]  (memory-bound GEMV)
__global__ __launch_bounds__(256) void outvec_k(const float* __restrict__ text,
                                                const float* __restrict__ weight,
                                                float* __restrict__ out)
{
  const int b = blockIdx.x >> 2;
  const int hb = blockIdx.x & 3;
  const int t = threadIdx.x;
  __shared__ float w[S_];
  w[t]       = weight[b * S_ + t];
  w[t + 256] = weight[b * S_ + 256 + t];
  __syncthreads();
  const int h = hb * 256 + t;
  const float* tb = text + (size_t)b * S_ * H_ + h;
  float acc = 0.0f;
#pragma unroll 4
  for (int s = 0; s < S_; ++s) acc += tb[(size_t)s * H_] * w[s];
  out[(size_t)b * H_ + h] = acc;
}

extern "C" void kernel_launch(void* const* d_in, const int* in_sizes, int n_in,
                              void* d_out, int out_size, void* d_ws, size_t ws_size,
                              hipStream_t stream) {
  (void)in_sizes; (void)n_in; (void)out_size; (void)ws_size;
  const float* text      = (const float*)d_in[0];  // [B,S,H]
  const float* aspect    = (const float*)d_in[1];  // [B,S,E]
  const float* w_text    = (const float*)d_in[2];  // [B,H,H]
  const float* w_aspect  = (const float*)d_in[3];  // [B,E,E]
  const float* w_combine = (const float*)d_in[4];  // [B,1,H+E]

  float* weight = (float*)d_out;            // [B,S]
  float* out    = (float*)d_out + B_ * S_;  // [B,H]
  float* scores = (float*)d_ws;             // [B,S] f32 scratch

  zero_f32<<<(B_ * S_ + 255) / 256, 256, 0, stream>>>(scores, B_ * S_);

  // m_text contribution: R=1024 (8 row blocks), K=1024, wc offset 0
  gemm_tanh_score<<<B_ * 8 * 4, 256, 0, stream>>>(
      w_text, text, w_combine, scores, H_, 8, 4, 0);
  // m_aspect contribution: R=768 (6 row blocks), K=768, wc offset H
  gemm_tanh_score<<<B_ * 6 * 4, 256, 0, stream>>>(
      w_aspect, aspect, w_combine, scores, E_, 6, 4, H_);

  softmax_k<<<B_, 256, 0, stream>>>(scores, weight);
  outvec_k<<<B_ * 4, 256, 0, stream>>>(text, weight, out);
}
